// GCN_83004537962756
// MI455X (gfx1250) — matrix operand order repaired and checked by reading, not compiled, not run on hardware
//
#include <hip/hip_runtime.h>
#include <hip/hip_bf16.h>

#define NNODES 100000
#define NEDGES 3200000
#define INDIM  128
#define HID    64
#define NG     256

typedef __attribute__((ext_vector_type(16))) __bf16 v16bf;
typedef __attribute__((ext_vector_type(8)))  float  v8f;

static inline size_t alignup(size_t x) { return (x + 255) & ~size_t(255); }

// ---------------- utility fills ----------------
__global__ void k_fill(float* __restrict__ p, float v, int n) {
    int i = blockIdx.x * blockDim.x + threadIdx.x;
    if (i < n) p[i] = v;
}

// ---------------- degree accumulation (self-loops pre-seeded as deg=1) ----------------
__global__ void k_deg(const int* __restrict__ col, const float* __restrict__ ew,
                      float* __restrict__ deg, int e) {
    int i = blockIdx.x * blockDim.x + threadIdx.x;
    if (i < e) atomicAdd(&deg[col[i]], ew[i]);
}

// deg -> dis = deg>0 ? rsqrt(deg) : 0   (in place)
__global__ void k_dis(float* __restrict__ deg, int n) {
    int i = blockIdx.x * blockDim.x + threadIdx.x;
    if (i < n) {
        float d = deg[i];
        deg[i] = (d > 0.0f) ? rsqrtf(fmaxf(d, 1e-30f)) : 0.0f;
    }
}

__global__ void k_norm(const int* __restrict__ row, const int* __restrict__ col,
                       const float* __restrict__ ew, const float* __restrict__ dis,
                       float* __restrict__ norm, int e) {
    int i = blockIdx.x * blockDim.x + threadIdx.x;
    if (i < e) norm[i] = dis[row[i]] * ew[i] * dis[col[i]];
}

// ---------------- WMMA GEMM: Y[nrows,64] = X[nrows,KDIM] @ W[KDIM,64] ----------------
// 128 threads = 4 waves; wave w -> 16-col tile. Block sweeps ROWTILES 16-row tiles,
// keeping ALL B fragments (whole W column tile) resident in VGPRs.
// Requires nrows % 16 == 0 (true: N=100000), tile skip is wave-uniform -> EXEC all-1s.
template <int KDIM, int ROWTILES>
__global__ void __launch_bounds__(128)
k_gemm_wmma(const float* __restrict__ X, const float* __restrict__ W,
            float* __restrict__ Y, int nrows) {
    constexpr int KT = KDIM / 32;
    const int lane = threadIdx.x & 31;
    const int wave = threadIdx.x >> 5;     // 0..3 -> column tile
    const int m    = lane & 15;
    const int hi   = lane >> 4;            // lane half selects K-group
    const int ncol = wave * 16 + m;        // B/D column for this lane

    // ---- B fragments (32x16 bf16 each), hoisted across all row tiles ----
    v16bf bfrag[KT];
#pragma unroll
    for (int kt = 0; kt < KT; ++kt) {
        const float* wp = W + (size_t)(kt * 32 + hi * 16) * HID + ncol;
#pragma unroll
        for (int j = 0; j < 8; ++j) {
            bfrag[kt][2 * j]     = (__bf16)wp[(2 * j) * HID];
            bfrag[kt][2 * j + 1] = (__bf16)wp[(2 * j + 1) * HID];
        }
    }

#pragma unroll
    for (int t = 0; t < ROWTILES; ++t) {
        const int rowbase = (blockIdx.x * ROWTILES + t) * 16;
        if (rowbase >= nrows) break;       // uniform across block

        v8f acc = {};
        const float* xp = X + (size_t)(rowbase + m) * KDIM;
#pragma unroll
        for (int kt = 0; kt < KT; ++kt) {
            // A fragment (16x32 bf16), contiguous k = 16*hi + s, same order as B
            v16bf a;
#pragma unroll
            for (int j = 0; j < 8; ++j) {
                int kb = kt * 32 + hi * 16 + j * 2;
                a[2 * j]     = (__bf16)xp[kb];
                a[2 * j + 1] = (__bf16)xp[kb + 1];
            }
            acc = __builtin_amdgcn_wmma_f32_16x16x32_bf16(
                false, a, false, bfrag[kt], (short)0, acc, false, false);
        }
        // D store: VGPR r -> row rowbase + r + 8*hi, col = ncol (tile always full)
        float* yp = Y + (size_t)(rowbase + hi * 8) * HID + ncol;
#pragma unroll
        for (int r = 0; r < 8; ++r) yp[(size_t)r * HID] = acc[r];
    }
}

// ---------------- edge scatter: h[col] += xl[row]*norm ; self-loops appended ----------------
// one WAVE per edge: lane handles features (2*lane, 2*lane+1) -> float2 coalesced gather,
// edge metadata broadcast via readfirstlane so it lowers to scalar loads.
__global__ void __launch_bounds__(256)
k_scatter(const int* __restrict__ row, const int* __restrict__ col,
          const float* __restrict__ norm, const float* __restrict__ dis,
          const float* __restrict__ xl, float* __restrict__ h, int e, int n) {
    int wid = blockIdx.x * (blockDim.x >> 5) + (threadIdx.x >> 5);
    wid = __builtin_amdgcn_readfirstlane(wid);      // wave-uniform edge id
    if (wid >= e + n) return;
    const int lane = threadIdx.x & 31;
    int r, c; float w;
    if (wid < e) {
        r = row[wid]; c = col[wid]; w = norm[wid];
    } else {
        int i = wid - e;
        r = i; c = i;
        float d = dis[i];
        w = d * d;                                   // self-loop: weight 1 * dis^2
    }
    const float2 v = *(const float2*)(xl + (size_t)r * HID + 2 * lane);
    float* hp = h + (size_t)c * HID + 2 * lane;
    atomicAdd(hp,     v.x * w);
    atomicAdd(hp + 1, v.y * w);
}

__global__ void k_bias_relu(float* __restrict__ hbuf, const float* __restrict__ b, int total) {
    int t = blockIdx.x * blockDim.x + threadIdx.x;
    if (t < total) {
        int f = t & (HID - 1);
        hbuf[t] = fmaxf(hbuf[t] + b[f], 0.0f);
    }
}

// ---------------- mean pool accumulation: one wave per node ----------------
__global__ void __launch_bounds__(256)
k_pool(const int* __restrict__ batch, const float* __restrict__ hbuf,
       float* __restrict__ sums, float* __restrict__ cnt, int n) {
    int i = blockIdx.x * (blockDim.x >> 5) + (threadIdx.x >> 5);
    i = __builtin_amdgcn_readfirstlane(i);
    if (i >= n) return;
    const int lane = threadIdx.x & 31;
    const int g = batch[i];
    const float2 v = *(const float2*)(hbuf + (size_t)i * HID + 2 * lane);
    float* sp = sums + (size_t)g * HID + 2 * lane;
    atomicAdd(sp,     v.x);
    atomicAdd(sp + 1, v.y);
    if (lane == 0) atomicAdd(&cnt[g], 1.0f);
}

// ---------------- final classifier: out[g] = mean(h_g) @ Wc + bc ----------------
__global__ void k_final(const float* __restrict__ sums, const float* __restrict__ cnt,
                        const float* __restrict__ Wc, const float* __restrict__ bc,
                        float* __restrict__ out) {
    int g = blockIdx.x * blockDim.x + threadIdx.x;
    if (g < NG) {
        float inv = 1.0f / fmaxf(cnt[g], 1.0f);
        float s = 0.0f;
#pragma unroll
        for (int f = 0; f < HID; ++f) s += sums[(size_t)g * HID + f] * Wc[f];
        out[g] = s * inv + bc[0];
    }
}

extern "C" void kernel_launch(void* const* d_in, const int* in_sizes, int n_in,
                              void* d_out, int out_size, void* d_ws, size_t ws_size,
                              hipStream_t stream) {
    const float* x     = (const float*)d_in[0];
    const int*   ei    = (const int*)d_in[1];     // [2,E] flat: row=ei[0:E], col=ei[E:2E]
    const float* ew    = (const float*)d_in[2];
    const int*   batch = (const int*)d_in[3];
    const float* W1    = (const float*)d_in[4];
    const float* b1    = (const float*)d_in[5];
    const float* W2    = (const float*)d_in[6];
    const float* b2    = (const float*)d_in[7];
    const float* Wc    = (const float*)d_in[8];
    const float* bc    = (const float*)d_in[9];
    float* out = (float*)d_out;

    const int N = NNODES, E = NEDGES;
    const int* row = ei;
    const int* col = ei + E;

    // ---- workspace carve-up ----
    char* ws = (char*)d_ws;
    size_t o = 0;
    float* deg  = (float*)(ws + o); o += alignup((size_t)N * 4);          // deg -> dis in place
    float* nrm  = (float*)(ws + o); o += alignup((size_t)E * 4);
    float* B0   = (float*)(ws + o); o += alignup((size_t)N * HID * 4);    // xl / h1l
    float* B1   = (float*)(ws + o); o += alignup((size_t)N * HID * 4);    // h1 / h2 (scatter dst)
    float* sums = (float*)(ws + o); o += alignup((size_t)NG * HID * 4);
    float* cnt  = (float*)(ws + o); o += alignup((size_t)NG * 4);

    const int T = 256;
    auto blks = [](long long total, int t) { return (unsigned)((total + t - 1) / t); };

    const int ETOT = E + N;                      // edges + self loops
    const unsigned edgeBlocks = blks(ETOT, 8);   // 8 waves/block, wave per edge
    const unsigned nodeBlocks = blks(N, 8);
    const unsigned gemmBlocks = blks((N + 15) / 16, 4);  // 4 row tiles per block

    // 1) normalization
    k_fill<<<blks(N, T), T, 0, stream>>>(deg, 1.0f, N);                 // self-loop weight
    k_deg<<<blks(E, T), T, 0, stream>>>(col, ew, deg, E);
    k_dis<<<blks(N, T), T, 0, stream>>>(deg, N);                        // deg -> dis
    k_norm<<<blks(E, T), T, 0, stream>>>(row, col, ew, deg, nrm, E);

    // 2) layer 1: xl = x @ W1 (WMMA), h1 = relu(scatter + b1)
    k_gemm_wmma<INDIM, 4><<<gemmBlocks, 128, 0, stream>>>(x, W1, B0, N);
    k_fill<<<blks((long long)N * HID, T), T, 0, stream>>>(B1, 0.0f, N * HID);
    k_scatter<<<edgeBlocks, T, 0, stream>>>(row, col, nrm, deg, B0, B1, E, N);
    k_bias_relu<<<blks((long long)N * HID, T), T, 0, stream>>>(B1, b1, N * HID);

    // 3) layer 2: h1l = h1 @ W2 (WMMA), h2 = relu(scatter + b2)
    k_gemm_wmma<HID, 4><<<gemmBlocks, 128, 0, stream>>>(B1, W2, B0, N);
    k_fill<<<blks((long long)N * HID, T), T, 0, stream>>>(B1, 0.0f, N * HID);
    k_scatter<<<edgeBlocks, T, 0, stream>>>(row, col, nrm, deg, B0, B1, E, N);
    k_bias_relu<<<blks((long long)N * HID, T), T, 0, stream>>>(B1, b2, N * HID);

    // 4) mean pool + classifier
    k_fill<<<blks(NG * HID + NG, T), T, 0, stream>>>(sums, 0.0f, NG * HID + NG); // sums+cnt contiguous
    k_pool<<<nodeBlocks, T, 0, stream>>>(batch, B1, sums, cnt, N);
    k_final<<<1, NG, 0, stream>>>(sums, cnt, Wc, bc, out);
}
